// Attention_87076166960040
// MI455X (gfx1250) — matrix operand order repaired and checked by reading, not compiled, not run on hardware
//
#include <hip/hip_runtime.h>

// ---------------------------------------------------------------------------
// CDNA5 wave32 WMMA attention (B=8, NQ=NKV=1024, D=512, H=8, DH=64)
// Memory-bound problem (~580MB min HBM traffic @ 23.3TB/s) -> bf16 WMMA path,
// all matmuls on v_wmma_f32_16x16x32_bf16, fp32 accumulation.
// R (256MB) is the dominant stream: staged into LDS in d-chunks via gfx1250
// async global->LDS DMA (ASYNCcnt + s_wait_asynccnt) so staging never touches
// VGPRs; falls back to load+ds_store if the builtins are absent.
// ---------------------------------------------------------------------------

typedef __attribute__((ext_vector_type(16))) __bf16 v16bf;
typedef __attribute__((ext_vector_type(8)))  float  v8f;

#define WMMA_BF16(a, b, c) \
  __builtin_amdgcn_wmma_f32_16x16x32_bf16(false, (a), false, (b), (short)0, (c), false, false)

#if defined(__gfx1250__) &&                                             \
    __has_builtin(__builtin_amdgcn_global_load_async_to_lds_b128) &&    \
    __has_builtin(__builtin_amdgcn_s_wait_asynccnt)
#define USE_ASYNC_LDS 1
// Exact parameter types per hipcc diagnostic: int4 (vector_size(16)) pointers,
// param0 in the __device__ (global, AS1) space, param1 the LDS (AS3) side.
typedef int v4i_ __attribute__((vector_size(16)));
typedef __attribute__((address_space(1))) v4i_* as1_v4i;
typedef __attribute__((address_space(3))) v4i_* as3_v4i;
#define ASYNC_G2L_B128(gsrc, ldst)                                       \
  __builtin_amdgcn_global_load_async_to_lds_b128(                        \
      (as1_v4i)(unsigned long long)(gsrc),                               \
      (as3_v4i)(unsigned int)(unsigned long long)(ldst), 0, 0)
#else
#define USE_ASYNC_LDS 0
#endif

// A-fragment (16x32 bf16) element->K mapping for a lane:
//   M=lane&15, K = hi*16 + e (contiguous, matches B fills), hi = lane>>4
__device__ __forceinline__ int a_kidx(int e, int hi) {
  return hi * 16 + e;
}

static constexpr int Bb  = 8;
static constexpr int NQ  = 1024;
static constexpr int NKV = 1024;
static constexpr int H   = 8;
static constexpr int DH  = 64;
static constexpr int HD  = 512;   // H*DH
static constexpr float SCALE = 0.125f;  // DH^-0.5

// ---------------------------------------------------------------------------
// Kernel 1/4: C[M,N] = A[M,K] @ W[K,N] + bias  (A,W fp32 in; OutT = __bf16 or float)
// 64x64 block tile, 8 waves (4x2), each wave owns a 16x32 D tile -> 2 WMMAs/k-step.
// ---------------------------------------------------------------------------
template <typename OutT>
__global__ __launch_bounds__(256) void gemm_wmma(const float* __restrict__ A,
                                                 const float* __restrict__ W,
                                                 const float* __restrict__ bias,
                                                 OutT* __restrict__ C,
                                                 int M, int N, int K) {
  constexpr int BM = 64, BN = 64, BK = 32;
  __shared__ __bf16 As[BM][BK + 2];
  __shared__ __bf16 Ws[BK][BN + 2];

  const int tid  = threadIdx.x;
  const int lane = tid & 31;
  const int wav  = tid >> 5;
  const int hi   = lane >> 4, l15 = lane & 15;
  const int wr   = wav & 3;        // wave row   (4)
  const int wc   = wav >> 2;       // wave col   (2), each covers 32 cols
  const int mBase = blockIdx.x * BM;
  const int nBase = blockIdx.y * BN;

  v8f acc0 = {}, acc1 = {};

  for (int k0 = 0; k0 < K; k0 += BK) {
    for (int i = tid; i < BM * BK; i += 256) {
      int r = i / BK, c = i % BK;
      As[r][c] = (__bf16)A[(size_t)(mBase + r) * K + (k0 + c)];
    }
    for (int i = tid; i < BK * BN; i += 256) {
      int r = i / BN, c = i % BN;
      Ws[r][c] = (__bf16)W[(size_t)(k0 + r) * N + (nBase + c)];
    }
    __syncthreads();

    v16bf a, b0, b1;
    const int arow = wr * 16 + l15;
#pragma unroll
    for (int e = 0; e < 16; ++e) a[e] = As[arow][a_kidx(e, hi)];
#pragma unroll
    for (int e = 0; e < 16; ++e) {
      int kk = hi * 16 + e;                 // B (32x16): lane=col, K striped by half
      b0[e] = Ws[kk][wc * 32 + l15];
      b1[e] = Ws[kk][wc * 32 + 16 + l15];
    }
    acc0 = WMMA_BF16(a, b0, acc0);
    acc1 = WMMA_BF16(a, b1, acc1);
    __syncthreads();
  }

  // D layout: VGPR i -> row hi*8+i, col l15
#pragma unroll
  for (int i = 0; i < 8; ++i) {
    int r  = mBase + wr * 16 + hi * 8 + i;
    int c0 = nBase + wc * 32 + l15;
    C[(size_t)r * N + c0]      = (OutT)(acc0[i] + bias[c0]);
    C[(size_t)r * N + c0 + 16] = (OutT)(acc1[i] + bias[c0 + 16]);
  }
}

// ---------------------------------------------------------------------------
// Kernel 2: scores = (Q.K^T + Q.R) * scale, mask, online max/sum.
// Block = (qtile of 16, batch). Wave w = head w. Raw scores -> attn buffer.
// Bias is seeded straight into the WMMA C-fragment (lane owns rows hi*8+i, col
// l15 -- exactly the D/C layout). R tile is cooperatively staged through LDS in
// 16-deep d-chunks (async DMA when available): bounds VGPR liveness and
// de-duplicates the 8 heads' reads.
// ---------------------------------------------------------------------------
__global__ __launch_bounds__(256) void scores_kernel(const __bf16* __restrict__ Qb,
                                                     const __bf16* __restrict__ Kb,
                                                     const float* __restrict__ R,
                                                     const int* __restrict__ mask,
                                                     float* __restrict__ Sout,
                                                     float* __restrict__ Mrow,
                                                     float* __restrict__ Lrow) {
  __shared__ float Qf[H][16][DH + 1];                 // fp32 Q tile, all heads (~33KB)
  __shared__ __attribute__((aligned(16))) float Rt[16][16][24];  // R d-chunk tile (~24KB)

  const int tid  = threadIdx.x;
  const int lane = tid & 31;
  const int h    = tid >> 5;            // wave == head
  const int hi   = lane >> 4, l15 = lane & 15;
  const int q0   = blockIdx.x * 16;
  const int b    = blockIdx.y;

  for (int i = tid; i < H * 16 * DH; i += 256) {
    int d = i & 63, t = i >> 6, qq = t & 15, hh = t >> 4;
    Qf[hh][qq][d] = (float)Qb[((size_t)(b * NQ + q0 + qq)) * HD + hh * DH + d];
  }
  __syncthreads();

  // Q A-fragments (K-dim = 64 -> 2 chunks of 32)
  v16bf aq0, aq1;
#pragma unroll
  for (int e = 0; e < 16; ++e) {
    aq0[e] = (__bf16)Qf[h][l15][a_kidx(e, hi)];
    aq1[e] = (__bf16)Qf[h][l15][32 + a_kidx(e, hi)];
  }

  float m[8], l[8];
#pragma unroll
  for (int i = 0; i < 8; ++i) { m[i] = -__builtin_inff(); l[i] = 0.f; }

  for (int k0 = 0; k0 < NKV; k0 += 16) {
    // Prefetch next R tile row-starts (global_prefetch_b8) while we chew this one.
    if (k0 + 16 < NKV) {
      const float* nxt =
          R + ((size_t)(q0 + (tid >> 4)) * NKV + (k0 + 16 + (tid & 15))) * DH;
      __builtin_prefetch(nxt, 0, 1);
    }

    // --- RPE bias accumulated over 4 LDS-staged d-chunks of 16:
    //     cacc[i] = Q[h, q=hi*8+i, :] . R[q0+q, k0+l15, :]
    float cacc[8];
#pragma unroll
    for (int i = 0; i < 8; ++i) cacc[i] = 0.f;

    for (int dc = 0; dc < DH; dc += 16) {
      __syncthreads();  // all readers of previous Rt chunk done
      for (int i = tid; i < 16 * 16 * 4; i += 256) {   // 1024 x 16B chunks
        int d4 = i & 3, kk = (i >> 2) & 15, qq = i >> 6;
        const float* gsrc =
            R + ((size_t)(q0 + qq) * NKV + (k0 + kk)) * DH + dc + d4 * 4;
        float* ldst = &Rt[qq][kk][d4 * 4];
#if USE_ASYNC_LDS
        // gfx1250 DMA: global -> LDS without touching VGPRs (ASYNCcnt-tracked)
        ASYNC_G2L_B128(gsrc, ldst);
#else
        *(float4*)ldst = *(const float4*)gsrc;
#endif
      }
#if USE_ASYNC_LDS
      __builtin_amdgcn_s_wait_asynccnt(0);
#endif
      __syncthreads();
#pragma unroll
      for (int i = 0; i < 8; ++i) {
        int q = hi * 8 + i;
        float s = 0.f;
#pragma unroll
        for (int dd = 0; dd < 16; ++dd)
          s += Qf[h][q][dc + dd] * Rt[q][l15][dd];
        cacc[i] += s;
      }
    }
    v8f c;
#pragma unroll
    for (int i = 0; i < 8; ++i) c[i] = cacc[i];

    // --- K^T B-fragments: col = key (l15), K-dim = d (contiguous 16 per lane)
    const __bf16* krow = Kb + ((size_t)(b * NKV + k0 + l15)) * HD + h * DH;
    v16bf bk0, bk1;
#pragma unroll
    for (int e = 0; e < 16; ++e) {
      bk0[e] = krow[hi * 16 + e];
      bk1[e] = krow[32 + hi * 16 + e];
    }
    c = WMMA_BF16(aq0, bk0, c);
    c = WMMA_BF16(aq1, bk1, c);

    // --- scale + mask, online stats, raw-score store
    const int kcol = k0 + l15;
    const int mk   = mask[b * NKV + kcol];
#pragma unroll
    for (int i = 0; i < 8; ++i) {
      float s = c[i] * SCALE;
      s = (mk == 0) ? -__builtin_inff() : s;

      float tmax = s;
#pragma unroll
      for (int off = 1; off < 16; off <<= 1)
        tmax = fmaxf(tmax, __shfl_xor(tmax, off, 32));
      float mnew = fmaxf(m[i], tmax);
      float p    = (s == -__builtin_inff()) ? 0.f : __expf(s - mnew);
      float psum = p;
#pragma unroll
      for (int off = 1; off < 16; off <<= 1)
        psum += __shfl_xor(psum, off, 32);
      float corr = (m[i] == -__builtin_inff()) ? 0.f : __expf(m[i] - mnew);
      l[i] = l[i] * corr + psum;
      m[i] = mnew;

      Sout[((size_t)(b * H + h) * NQ + (q0 + hi * 8 + i)) * NKV + kcol] = s;
    }
  }

  if (l15 == 0) {
#pragma unroll
    for (int i = 0; i < 8; ++i) {
      int q = q0 + hi * 8 + i;
      Mrow[(size_t)(b * H + h) * NQ + q] = m[i];
      Lrow[(size_t)(b * H + h) * NQ + q] = l[i];
    }
  }
}

// ---------------------------------------------------------------------------
// Kernel 3: attn = exp(S-m)/l (written in place, final output) and O = attn @ V.
// Block = (qtile, batch); wave = head. k-steps of 32: 1 A-frag (P), 4 B-frags (V),
// 4 WMMAs into the 16x64 fp32 accumulator. O written in "bqhd" layout.
// ---------------------------------------------------------------------------
__global__ __launch_bounds__(256) void pv_kernel(float* __restrict__ Sattn,
                                                 const __bf16* __restrict__ Vb,
                                                 const float* __restrict__ Mrow,
                                                 const float* __restrict__ Lrow,
                                                 float* __restrict__ O) {
  const int tid  = threadIdx.x;
  const int lane = tid & 31;
  const int h    = tid >> 5;
  const int hi   = lane >> 4, l15 = lane & 15;
  const int q0   = blockIdx.x * 16;
  const int b    = blockIdx.y;

  const size_t rowStat = (size_t)(b * H + h) * NQ + q0 + l15;
  const float mq   = Mrow[rowStat];
  const float lq   = Lrow[rowStat];
  const float invl = (lq > 0.f) ? 1.f / lq : 0.f;

  v8f acc[4] = {};

  for (int k0 = 0; k0 < NKV; k0 += 32) {
    // A-fragment = P[16 q, 32 k]; finalize attn in place while loading
    float* srow = Sattn + ((size_t)(b * H + h) * NQ + (q0 + l15)) * NKV + k0;
    v16bf ap;
#pragma unroll
    for (int e = 0; e < 16; ++e) {
      int kk  = a_kidx(e, hi);   // both halves cover disjoint k in [0,32)
      float s = srow[kk];
      float p = (s == -__builtin_inff()) ? 0.f : __expf(s - mq) * invl;
      srow[kk] = p;
      ap[e]    = (__bf16)p;
    }
    // B-fragments = V[32 k, 64 d] -> 4 column subtiles
#pragma unroll
    for (int sub = 0; sub < 4; ++sub) {
      v16bf bv;
#pragma unroll
      for (int e = 0; e < 16; ++e) {
        int key = k0 + hi * 16 + e;
        bv[e] = Vb[((size_t)(b * NKV + key)) * HD + h * DH + sub * 16 + l15];
      }
      acc[sub] = WMMA_BF16(ap, bv, acc[sub]);
    }
  }

#pragma unroll
  for (int sub = 0; sub < 4; ++sub)
#pragma unroll
    for (int i = 0; i < 8; ++i) {
      int q = q0 + hi * 8 + i;
      O[((size_t)(b * NQ + q)) * HD + h * DH + sub * 16 + l15] = acc[sub][i];
    }
}

// ---------------------------------------------------------------------------
// Launch
// ---------------------------------------------------------------------------
extern "C" void kernel_launch(void* const* d_in, const int* in_sizes, int n_in,
                              void* d_out, int out_size, void* d_ws, size_t ws_size,
                              hipStream_t stream) {
  const float* q    = (const float*)d_in[0];
  const float* kv   = (const float*)d_in[1];
  const int*   mask = (const int*)d_in[2];
  const float* Wq   = (const float*)d_in[3];
  const float* bq   = (const float*)d_in[4];
  const float* Wk   = (const float*)d_in[5];
  const float* bk   = (const float*)d_in[6];
  const float* Wv   = (const float*)d_in[7];
  const float* bv   = (const float*)d_in[8];
  const float* Wo   = (const float*)d_in[9];
  const float* bo   = (const float*)d_in[10];
  const float* R    = (const float*)d_in[11];

  float* out  = (float*)d_out;                               // [B, NQ, 512]
  float* attn = (float*)d_out + (size_t)Bb * NQ * HD;        // [B, H, NQ, NKV]

  const size_t PROJ = (size_t)Bb * NQ * HD;                  // 4M elements
  char* ws = (char*)d_ws;
  __bf16* Qb   = (__bf16*)(ws);
  __bf16* Kb   = (__bf16*)(ws + PROJ * 2);
  __bf16* Vb   = (__bf16*)(ws + PROJ * 4);
  float*  Obuf = (float*)(ws + PROJ * 6);
  float*  Mrow = (float*)(ws + PROJ * 6 + PROJ * 4);
  float*  Lrow = Mrow + (size_t)Bb * H * NQ;

  const int M = Bb * NQ;          // 8192
  dim3 gGemm(M / 64, HD / 64);    // (128, 8)
  dim3 blk(256);

  // 1) projections -> bf16 Q/K/V
  gemm_wmma<__bf16><<<gGemm, blk, 0, stream>>>(q,  Wq, bq, Qb, M, HD, HD);
  gemm_wmma<__bf16><<<gGemm, blk, 0, stream>>>(kv, Wk, bk, Kb, M, HD, HD);
  gemm_wmma<__bf16><<<gGemm, blk, 0, stream>>>(kv, Wv, bv, Vb, M, HD, HD);

  // 2) scores + RPE bias + mask + online softmax stats (raw S -> attn buffer)
  //    grid x = qtile so all batches of one qtile share R slice via L2
  scores_kernel<<<dim3(NQ / 16, Bb), blk, 0, stream>>>(Qb, Kb, R, mask, attn, Mrow, Lrow);

  // 3) finalize attn in place + O = attn @ V
  pv_kernel<<<dim3(NQ / 16, Bb), blk, 0, stream>>>(attn, Vb, Mrow, Lrow, Obuf);

  // 4) out = O @ Wo + bo (fp32 output)
  gemm_wmma<float><<<gGemm, blk, 0, stream>>>(Obuf, Wo, bo, out, M, HD, HD);
}
